// SynaptoGenesis_59871844106394
// MI455X (gfx1250) — matrix operand order repaired and checked by reading, not compiled, not run on hardware
//
#include <hip/hip_runtime.h>

typedef __attribute__((ext_vector_type(16))) _Float16 v16h;
typedef __attribute__((ext_vector_type(8)))  _Float16 v8h;
typedef __attribute__((ext_vector_type(8)))  float    v8f;

#define N_NODES 8192
#define N_EDGES 16384
#define DIM     64
#define PCOLS   64                    // columns staged per LDS panel (8 KB)
#define NEGV    (-1.0e10f)
#define FNINF   (-3.0e38f)
#define LN2     0.69314718056f

__device__ __forceinline__ unsigned hash_u32(unsigned x) {
    x ^= x >> 16; x *= 0x7feb352dU;
    x ^= x >> 15; x *= 0x846ca68bU;
    x ^= x >> 16; return x;
}
__device__ __forceinline__ float u01(unsigned x) {
    return (float)(hash_u32(x) >> 8) * (1.0f / 16777216.0f) + 1e-7f;
}

// ---------------------------------------------------------------------------
// Kernel 1: per-node prep — probs/gens, f16 nodes, hninv = act?rsqrt(|h|^2):0,
// hbias = act?0:NEG (folds the active-mask + norm into 1 mul + 1 add later)
// ---------------------------------------------------------------------------
__global__ __launch_bounds__(256)
void prep_kernel(const float* __restrict__ nodes, const float* __restrict__ Wp,
                 const float* __restrict__ bp, const float* __restrict__ active_nodes,
                 _Float16* __restrict__ nodes_f16, float* __restrict__ hninv,
                 float* __restrict__ hbias, float* __restrict__ gens) {
    int i = blockIdx.x * blockDim.x + threadIdx.x;
    if (i >= N_NODES) return;
    float accp = 0.f, h2 = 0.f;
    const float* row = nodes + (size_t)i * DIM;
    _Float16* drow = nodes_f16 + (size_t)i * DIM;
    #pragma unroll 8
    for (int d = 0; d < DIM; ++d) {
        float x = row[d];
        accp += x * Wp[d];
        h2 += x * x;
        drow[d] = (_Float16)x;
    }
    float a = active_nodes[i];
    hninv[i] = (a > 0.f) ? rsqrtf(fmaxf(h2, 1e-16f)) : 0.f;
    hbias[i] = (a > 0.f) ? 0.f : NEGV;
    float p = 1.f / (1.f + __expf(-(accp + bp[0])));
    float u = u01((unsigned)i * 2654435761u ^ 0x9E3779B9u);
    gens[i] = (u < p * a) ? 1.f : 0.f;
}

// ---------------------------------------------------------------------------
// Kernel 2: queries = nodes @ Wq + bq -> f16 + qn  (67 MFLOP, negligible)
// ---------------------------------------------------------------------------
__global__ __launch_bounds__(256)
void queries_kernel(const float* __restrict__ nodes, const float* __restrict__ Wq,
                    const float* __restrict__ bq, _Float16* __restrict__ qf16,
                    float* __restrict__ qn) {
    __shared__ float snode[4][DIM];
    __shared__ float sq[4][DIM];
    int t = threadIdx.x, r = t >> 6, col = t & 63;
    int row = blockIdx.x * 4 + r;
    snode[r][col] = nodes[(size_t)row * DIM + col];
    __syncthreads();
    float acc = bq[col];
    #pragma unroll 8
    for (int k = 0; k < DIM; ++k)
        acc += snode[r][k] * Wq[(size_t)k * DIM + col];
    qf16[(size_t)row * DIM + col] = (_Float16)acc;
    sq[r][col] = acc * acc;
    __syncthreads();
    for (int off = 32; off > 0; off >>= 1) {
        if (col < off) sq[r][col] += sq[r][col + off];
        __syncthreads();
    }
    if (col == 0) qn[row] = sq[r][0];
}

// ---------------------------------------------------------------------------
// Branch-free per-tile epilogue (~13 VALU + 2 trans per element):
//   score: mul, mul, med3, add;  raw: med3(a,b,BIG)==max;  RNG: 1 mul_lo;
//   Gumbel: 2x v_log (neg folded as src modifier) + 1 fma (const shift of a
//   Gumbel is argmax-invariant, so no ln2 rescale of the inner log needed).
// ---------------------------------------------------------------------------
template<bool DIAG>
__device__ __forceinline__ void tile_epilogue(const v8f& c, const float* rqinv,
                                              const unsigned* rk, float hic, float bic,
                                              unsigned ch, int col, int rowBase, int hi,
                                              float* best, float* raw, int* bidx) {
    #pragma unroll
    for (int v = 0; v < 8; ++v) {
        float s = c[v] * (rqinv[v] * hic);
        s = __builtin_amdgcn_fmed3f(s, -10000.f, 10000.f) + bic;
        raw[v] = __builtin_amdgcn_fmed3f(s, raw[v], 3.0e38f);   // == max, no canon
        if (DIAG) {
            int rowm = rowBase + v + hi * 8;
            if (col == rowm) s = NEGV;                          // no self-loops
        }
        unsigned x = ch * rk[v];                                // 1-mul per-elem RNG
        float u = (float)(x >> 8) * 0x1p-24f + 0x1p-26f;        // u in (0,1)
        float l1 = __builtin_amdgcn_logf(u);                    // log2(u) < 0
        float l2 = __builtin_amdgcn_logf(-l1);                  // neg is src-mod
        float val = __builtin_fmaf(-LN2, l2, s);                // s + Gumbel - C
        if (val > best[v]) { best[v] = val; bidx[v] = col; }
    }
}

// ---------------------------------------------------------------------------
// Kernel 3: fused scores GEMM + Gumbel-max argmax.
// Wave owns 16 rows (A pinned in regs); block double-buffers 64-column B
// panels + hninv/hbias side panels into LDS via global_load_async_to_lds_b128
// (ASYNCcnt), then 4 tiles/panel: 2x v_wmma_f32_16x16x32_f16 + slim epilogue.
// Diagonal handling hoisted to a *scalar* panel-level branch (readfirstlane)
// so 127/128 panels run a fully branch-free body.
// ---------------------------------------------------------------------------
template<bool DIAGP>
__device__ __forceinline__ void panel_body(const _Float16* shb, const float* shAuxB,
                                           int p, int rbU, const v16h& a0, const v16h& a1,
                                           const float* rqinv, const unsigned* rk,
                                           int m, int hi, int rowBase,
                                           float* best, float* raw, int* bidx) {
    #pragma unroll
    for (int tt = 0; tt < PCOLS / 16; ++tt) {
        int colBase = p * PCOLS + tt * 16;
        // B = nodes_tile^T from LDS: lane m -> node row, 32 B per K-half
        const v16h* bp16 = reinterpret_cast<const v16h*>(shb + (tt * 16 + m) * DIM);
        v16h b0 = bp16[hi];
        v16h b1 = bp16[2 + hi];
        v8f c = {};
        c = __builtin_amdgcn_wmma_f32_16x16x32_f16(false, a0, false, b0,
                                                   (short)0, c, false, false);
        c = __builtin_amdgcn_wmma_f32_16x16x32_f16(false, a1, false, b1,
                                                   (short)0, c, false, false);
        int col = colBase + m;
        float hic = shAuxB[tt * 16 + m];
        float bic = shAuxB[PCOLS + tt * 16 + m];
        unsigned ch = hash_u32((unsigned)col * 0x9E3779B1u);
        if (DIAGP && colBase == rbU)                    // scalar compare
            tile_epilogue<true >(c, rqinv, rk, hic, bic, ch, col, rowBase, hi,
                                 best, raw, bidx);
        else
            tile_epilogue<false>(c, rqinv, rk, hic, bic, ch, col, rowBase, hi,
                                 best, raw, bidx);
    }
}

__global__ __launch_bounds__(256)
void score_argmax_kernel(const _Float16* __restrict__ qf16,
                         const _Float16* __restrict__ hf16,
                         const float* __restrict__ qn,
                         const float* __restrict__ hninv,
                         const float* __restrict__ hbias,
                         int* __restrict__ select_out, float* __restrict__ rowmax_out) {
    __shared__ _Float16 sh[2][PCOLS * DIM];                // 2 x 8 KB B panels
    __shared__ float    shAux[2][2 * PCOLS];               // hninv | hbias panels
    const int t = threadIdx.x;
    const int lane = t & 31;
    const int wave = t >> 5;
    const int rowBase = (blockIdx.x * 8 + wave) * 16;
    const int rbU = __builtin_amdgcn_readfirstlane(rowBase);  // wave-uniform -> SGPR
    const int m  = lane & 15;
    const int hi = lane >> 4;

    // --- A matrix (16x64 f16) fixed in registers for the full column sweep ---
    const _Float16* qrow = qf16 + (size_t)(rowBase + m) * DIM;
    const v8h* q8 = reinterpret_cast<const v8h*>(qrow);
    v8h p0 = q8[2 * hi], p1 = q8[2 * hi + 1], p2 = q8[4 + 2 * hi], p3 = q8[5 + 2 * hi];
    v16h a0, a1;
    #pragma unroll
    for (int i = 0; i < 8; ++i) {
        a0[i] = p0[i]; a0[i + 8] = p1[i];
        a1[i] = p2[i]; a1[i + 8] = p3[i];
    }

    float rqinv[8]; unsigned rk[8];
    #pragma unroll
    for (int v = 0; v < 8; ++v) {
        int rowm = rowBase + v + hi * 8;
        rqinv[v] = rsqrtf(fmaxf(qn[rowm], 1e-16f));
        rk[v] = hash_u32((unsigned)rowm * 9781u) | 1u;
    }

    float best[8], raw[8]; int bidx[8];
    #pragma unroll
    for (int v = 0; v < 8; ++v) { best[v] = FNINF; raw[v] = FNINF; bidx[v] = 0; }

    const unsigned ldsB   = (unsigned)(unsigned long long)(&sh[0][0]);
    const unsigned ldsAux = (unsigned)(unsigned long long)(&shAux[0][0]);
    const unsigned tOff = (unsigned)t * 32u;               // 8 KB / 256 thr
    const unsigned panelBytes = PCOLS * DIM * 2;           // 8192

    // issue async panel copy (B panel for all threads; aux panels lanes t<32)
    auto stage_panel = [&](int p, int dstBuf) {
        unsigned la = ldsB + (unsigned)dstBuf * panelBytes + tOff;
        unsigned go = (unsigned)p * panelBytes + tOff;
        asm volatile("global_load_async_to_lds_b128 %0, %1, %2\n\t"
                     "global_load_async_to_lds_b128 %0, %1, %2 offset:16"
                     :: "v"(la), "v"(go), "s"(hf16) : "memory");
        if (t < 16) {                                      // hninv: 256 B
            unsigned la2 = ldsAux + (unsigned)dstBuf * 512u + (unsigned)t * 16u;
            unsigned go2 = (unsigned)p * 256u + (unsigned)t * 16u;
            asm volatile("global_load_async_to_lds_b128 %0, %1, %2"
                         :: "v"(la2), "v"(go2), "s"(hninv) : "memory");
        } else if (t < 32) {                               // hbias: 256 B
            unsigned la2 = ldsAux + (unsigned)dstBuf * 512u + 256u
                         + (unsigned)(t - 16) * 16u;
            unsigned go2 = (unsigned)p * 256u + (unsigned)(t - 16) * 16u;
            asm volatile("global_load_async_to_lds_b128 %0, %1, %2"
                         :: "v"(la2), "v"(go2), "s"(hbias) : "memory");
        }
    };

    stage_panel(0, 0);                                     // preload panel 0

    const int NP = N_NODES / PCOLS;
    for (int p = 0; p < NP; ++p) {
        asm volatile("s_wait_asynccnt 0" ::: "memory");    // my slices landed
        __syncthreads();                                   // whole panel visible
        const int buf = p & 1;
        if (p + 1 < NP) stage_panel(p + 1, buf ^ 1);       // prefetch next

        const _Float16* shb = &sh[buf][0];
        const float* shAuxB = &shAux[buf][0];
        // scalar branch: only the panel containing this wave's rows needs the
        // per-element diagonal test
        if (rbU >= p * PCOLS && rbU < (p + 1) * PCOLS)
            panel_body<true >(shb, shAuxB, p, rbU, a0, a1, rqinv, rk,
                              m, hi, rowBase, best, raw, bidx);
        else
            panel_body<false>(shb, shAuxB, p, rbU, a0, a1, rqinv, rk,
                              m, hi, rowBase, best, raw, bidx);
    }

    // cross-lane argmax within the 16 lanes sharing each row (half-wave only)
    #pragma unroll
    for (int off = 1; off < 16; off <<= 1) {
        #pragma unroll
        for (int v = 0; v < 8; ++v) {
            float ob = __shfl_xor(best[v], off, 32);
            int   oi = __shfl_xor(bidx[v], off, 32);
            float orw = __shfl_xor(raw[v], off, 32);
            if (ob > best[v] || (ob == best[v] && oi < bidx[v])) {
                best[v] = ob; bidx[v] = oi;
            }
            raw[v] = __builtin_amdgcn_fmed3f(orw, raw[v], 3.0e38f);
        }
    }
    if ((lane & 15) == 0) {
        #pragma unroll
        for (int v = 0; v < 8; ++v) {
            int rowm = rowBase + v + hi * 8;
            select_out[rowm] = bidx[v];
            rowmax_out[rowm] = raw[v];
        }
    }
}

// ---------------------------------------------------------------------------
// Kernel 4: drop generators whose edge (i, select[i]) already exists
// ---------------------------------------------------------------------------
__global__ __launch_bounds__(256)
void exist_kernel(const int* __restrict__ receivers, const int* __restrict__ senders,
                  const float* __restrict__ active_edges, const int* __restrict__ sel,
                  float* __restrict__ gens) {
    int e = blockIdx.x * blockDim.x + threadIdx.x;
    if (e >= N_EDGES) return;
    if (active_edges[e] > 0.f) {
        int s = senders[e];
        if (sel[s] == receivers[e]) gens[s] = 0.f;
    }
}

// ---------------------------------------------------------------------------
// Kernel 5: single-block scans — cumsum(gens), cumsum(active_edges), scalars
// ---------------------------------------------------------------------------
__global__ __launch_bounds__(1024)
void scan_kernel(float* __restrict__ gens, const float* __restrict__ rowmax,
                 const float* __restrict__ active_edges,
                 float* __restrict__ gcum, float* __restrict__ acum,
                 float* __restrict__ scal) {
    __shared__ float sdata[1024];
    int t = threadIdx.x;

    float loc[8]; float s = 0.f;
    #pragma unroll
    for (int j = 0; j < 8; ++j) {
        int i = t * 8 + j;
        float g = gens[i];
        if (rowmax[i] <= FNINF) g = 0.f;
        gens[i] = g;
        s += g; loc[j] = s;
    }
    sdata[t] = s; __syncthreads();
    for (int off = 1; off < 1024; off <<= 1) {
        float v = (t >= off) ? sdata[t - off] : 0.f;
        __syncthreads();
        sdata[t] += v;
        __syncthreads();
    }
    float excl = sdata[t] - s;
    #pragma unroll
    for (int j = 0; j < 8; ++j) gcum[t * 8 + j] = excl + loc[j];
    float total_g = sdata[1023];
    __syncthreads();

    float loce[16]; s = 0.f;
    #pragma unroll
    for (int j = 0; j < 16; ++j) {
        s += active_edges[t * 16 + j];
        loce[j] = s;
    }
    sdata[t] = s; __syncthreads();
    for (int off = 1; off < 1024; off <<= 1) {
        float v = (t >= off) ? sdata[t - off] : 0.f;
        __syncthreads();
        sdata[t] += v;
        __syncthreads();
    }
    excl = sdata[t] - s;
    #pragma unroll
    for (int j = 0; j < 16; ++j) acum[t * 16 + j] = excl + loce[j];

    if (t == 0) {
        float e_active = sdata[1023];
        float allowed = (float)N_EDGES - e_active - 1.f;
        float n_g = fminf(fmaxf(total_g, 0.f), fmaxf(allowed, 0.f));
        scal[0] = e_active;
        scal[1] = n_g;
    }
}

// ---------------------------------------------------------------------------
// Kernel 6: naedges (_incr_edges closed form), mask_new, default nsend/nrec
// ---------------------------------------------------------------------------
__global__ __launch_bounds__(256)
void finalize_kernel(const float* __restrict__ active_edges,
                     const float* __restrict__ acum, const float* __restrict__ scal,
                     const int* __restrict__ receivers, const int* __restrict__ senders,
                     float* __restrict__ mask_new,
                     float* __restrict__ out_nsend, float* __restrict__ out_nrec,
                     float* __restrict__ out_na) {
    int e = blockIdx.x * blockDim.x + threadIdx.x;
    if (e >= N_EDGES) return;
    int n_g = (int)(scal[1] + 0.5f);
    float c = acum[e];
    float prev = (e - n_g - 1 >= 0) ? acum[e - n_g - 1] : 0.f;
    float na = ((c - prev) > 0.f) ? 1.f : 0.f;
    if (e == N_EDGES - 1) na = 0.f;
    float mn = na * (1.f - active_edges[e]);
    mask_new[e] = mn;
    float ns = (float)senders[e];
    float nr = (float)receivers[e];
    if (na <= 0.f) { ns = (float)(N_NODES - 1); nr = (float)(N_NODES - 1); }
    out_na[e] = na;
    out_nsend[e] = ns;
    out_nrec[e] = nr;
}

// ---------------------------------------------------------------------------
// Kernel 7: scatter new-edge endpoints (unique slots -> race-free)
// ---------------------------------------------------------------------------
__global__ __launch_bounds__(256)
void scatter_kernel(const float* __restrict__ gens, const float* __restrict__ gcum,
                    const int* __restrict__ sel, const float* __restrict__ mask_new,
                    const float* __restrict__ scal,
                    float* __restrict__ out_nsend, float* __restrict__ out_nrec) {
    int i = blockIdx.x * blockDim.x + threadIdx.x;
    if (i >= N_NODES) return;
    if (gens[i] > 0.f) {
        int e_act = (int)(scal[0] + 0.5f);
        int n_g   = (int)(scal[1] + 0.5f);
        int rank  = (int)(gcum[i] + 0.5f);
        if (rank >= 1 && rank <= n_g) {
            int slot = rank - 1 + e_act;
            if (slot < N_EDGES && mask_new[slot] > 0.f) {
                out_nsend[slot] = (float)i;
                out_nrec[slot]  = (float)sel[i];
            }
        }
    }
}

// ---------------------------------------------------------------------------
// Kernel 8: new_edges = edges + N(0,1)*mask_new  (native trans ops)
// ---------------------------------------------------------------------------
__global__ __launch_bounds__(256)
void edges_kernel(const float* __restrict__ edges, const float* __restrict__ mask_new,
                  float* __restrict__ out_edges) {
    int idx = blockIdx.x * blockDim.x + threadIdx.x;
    if (idx >= N_EDGES * DIM) return;
    float mn = mask_new[idx >> 6];
    float u1 = u01((unsigned)idx * 747796405u ^ 0x68E31DA4u);
    float u2 = u01((unsigned)idx * 2891336453u ^ 0x1B56C4E9u);
    float nl = __builtin_amdgcn_logf(u1) * (-2.f * LN2);   // -2 ln(u1)
    float z = __builtin_amdgcn_sqrtf(nl) * __cosf(6.28318530718f * u2);
    out_edges[idx] = edges[idx] + z * mn;
}

// ---------------------------------------------------------------------------
extern "C" void kernel_launch(void* const* d_in, const int* in_sizes, int n_in,
                              void* d_out, int out_size, void* d_ws, size_t ws_size,
                              hipStream_t stream) {
    const float* nodes        = (const float*)d_in[0];
    const float* edges        = (const float*)d_in[1];
    const int*   receivers    = (const int*)  d_in[2];
    const int*   senders      = (const int*)  d_in[3];
    const float* active_nodes = (const float*)d_in[4];
    const float* active_edges = (const float*)d_in[5];
    const float* Wq           = (const float*)d_in[6];
    const float* bq           = (const float*)d_in[7];
    const float* Wp           = (const float*)d_in[8];
    const float* bp           = (const float*)d_in[9];

    char* ws = (char*)d_ws;
    size_t off = 0;
    _Float16* nodes_f16 = (_Float16*)(ws + off); off += (size_t)N_NODES * DIM * 2;
    _Float16* qf16      = (_Float16*)(ws + off); off += (size_t)N_NODES * DIM * 2;
    float* qn       = (float*)(ws + off); off += N_NODES * 4;
    float* hninv    = (float*)(ws + off); off += N_NODES * 4;
    float* hbias    = (float*)(ws + off); off += N_NODES * 4;
    float* gens     = (float*)(ws + off); off += N_NODES * 4;
    int*   sel      = (int*)  (ws + off); off += N_NODES * 4;
    float* rowmax   = (float*)(ws + off); off += N_NODES * 4;
    float* gcum     = (float*)(ws + off); off += N_NODES * 4;
    float* acum     = (float*)(ws + off); off += N_EDGES * 4;
    float* mask_new = (float*)(ws + off); off += N_EDGES * 4;
    float* scal     = (float*)(ws + off); off += 256;

    float* out = (float*)d_out;
    float* out_edges = out;
    float* out_nsend = out + (size_t)N_EDGES * DIM;
    float* out_nrec  = out_nsend + N_EDGES;
    float* out_na    = out_nrec + N_EDGES;

    prep_kernel<<<N_NODES / 256, 256, 0, stream>>>(
        nodes, Wp, bp, active_nodes, nodes_f16, hninv, hbias, gens);

    queries_kernel<<<N_NODES / 4, 256, 0, stream>>>(
        nodes, Wq, bq, qf16, qn);

    score_argmax_kernel<<<N_NODES / 128, 256, 0, stream>>>(
        qf16, nodes_f16, qn, hninv, hbias, sel, rowmax);

    exist_kernel<<<N_EDGES / 256, 256, 0, stream>>>(
        receivers, senders, active_edges, sel, gens);

    scan_kernel<<<1, 1024, 0, stream>>>(
        gens, rowmax, active_edges, gcum, acum, scal);

    finalize_kernel<<<N_EDGES / 256, 256, 0, stream>>>(
        active_edges, acum, scal, receivers, senders,
        mask_new, out_nsend, out_nrec, out_na);

    scatter_kernel<<<N_NODES / 256, 256, 0, stream>>>(
        gens, gcum, sel, mask_new, scal, out_nsend, out_nrec);

    edges_kernel<<<(N_EDGES * DIM) / 256, 256, 0, stream>>>(
        edges, mask_new, out_edges);
}
